// MultiHeadAttention_67113158967390
// MI455X (gfx1250) — matrix operand order repaired and checked by reading, not compiled, not run on hardware
//
#include <hip/hip_runtime.h>

typedef _Float16 f16;
typedef __attribute__((ext_vector_type(16))) _Float16 v16h;
typedef __attribute__((ext_vector_type(8)))  _Float16 h8;
typedef __attribute__((ext_vector_type(8)))  float    v8f;
typedef __attribute__((ext_vector_type(4)))  float    f4;

#define D_MODEL   1024
#define NUM_HEADS 16
#define HEAD_DIM  64
#define SEQ       2048
#define BATCH     2
#define TOKENS    (BATCH * SEQ)            // 4096
#define QK_SCALE  0.125f                   // HEAD_DIM^-0.5 (exact in f16)
#define NEG_FILL  (-4.2949673e9f)          // module's -2^32+1

static __device__ __forceinline__ v8f wmma_f16(v16h a, v16h b, v8f c) {
  // D = A(16x32) x B(32x16) + C, f32 accumulate
  return __builtin_amdgcn_wmma_f32_16x16x32_f16(false, a, false, b, (short)0, c,
                                                false, false);
}

static __device__ __forceinline__ v8f zero8() {
  v8f z;
#pragma unroll
  for (int i = 0; i < 8; ++i) z[i] = 0.0f;
  return z;
}

// Fragment element e: e=0..7 <- p0[0..7], e=8..15 <- p1[0..7].
static __device__ __forceinline__ v16h frag_f16(const f16* p0, const f16* p1) {
  h8 a = *(const h8*)p0;
  h8 b = *(const h8*)p1;
  v16h r;
#pragma unroll
  for (int j = 0; j < 8; ++j) { r[j] = a[j]; r[8 + j] = b[j]; }
  return r;
}

// ---------------------------------------------------------------------------
// Kernel 0: one-shot f32 -> f16 conversion (bandwidth-bound, ~30 MB total).
// Removes per-WMMA cvt VALU work from all GEMM inner loops.
// ---------------------------------------------------------------------------
__global__ __launch_bounds__(256) void cvt_kernel(const float* __restrict__ src,
                                                  f16* __restrict__ dst, int n8) {
  const int i = blockIdx.x * blockDim.x + threadIdx.x;
  if (i < n8) {
    f4 a = *(const f4*)(src + (size_t)i * 8);
    f4 b = *(const f4*)(src + (size_t)i * 8 + 4);
    h8 o;
#pragma unroll
    for (int j = 0; j < 4; ++j) { o[j] = (f16)a[j]; o[4 + j] = (f16)b[j]; }
    *(h8*)(dst + (size_t)i * 8) = o;
  }
}

// ---------------------------------------------------------------------------
// QKV epilogue for one 16x16 C tile.
//   q -> Q[b][h][t][d] (f16, pre-scaled by 0.125)
//   k -> K[b][h][t][d] (f16)
//   v -> VT[b][h][d][t] (f16, transposed for the PV A-fragment)
// ---------------------------------------------------------------------------
static __device__ __forceinline__ void qkv_store(
    v8f acc, int mt, int nt, int m, int hf, const float* __restrict__ bias,
    f16* __restrict__ Q, f16* __restrict__ K, f16* __restrict__ VT) {
  const int o     = nt * 16 + m;         // lane's column
  const float bi  = bias[o];
  const int which = o >> 10;             // 0=q 1=k 2=v
  const int rem   = o & 1023;
  const int head  = rem >> 6;
  const int d     = rem & 63;
  const int tg0   = mt * 16 + hf * 8;    // first of lane's 8 token rows
  const int bb    = tg0 >> 11;
  const int t0    = tg0 & 2047;

  if (which == 2) {
    h8 ov;
#pragma unroll
    for (int j = 0; j < 8; ++j) ov[j] = (f16)(acc[j] + bi);
    size_t base = ((size_t)(bb * NUM_HEADS + head) * HEAD_DIM + d) * SEQ + t0;
    *(h8*)(VT + base) = ov;              // 8 consecutive t -> contiguous 16B
  } else {
    f16* dst      = (which == 0) ? Q : K;
    const float s = (which == 0) ? QK_SCALE : 1.0f;
    size_t base = ((size_t)(bb * NUM_HEADS + head) * SEQ + t0) * HEAD_DIM + d;
#pragma unroll
    for (int j = 0; j < 8; ++j)
      dst[base + (size_t)j * HEAD_DIM] = (f16)((acc[j] + bi) * s);
  }
}

// ---------------------------------------------------------------------------
// Kernel 1: fused QKV projection, 2x2 register-blocked (32x32 per wave).
// Pure f16 b128 loads + 4 WMMAs per K-step.
// ---------------------------------------------------------------------------
__global__ __launch_bounds__(256) void qkv_kernel(
    const f16* __restrict__ xh, const f16* __restrict__ wh,
    const float* __restrict__ bias,
    f16* __restrict__ Q, f16* __restrict__ K, f16* __restrict__ VT) {
  const int lane = threadIdx.x & 31;
  const int wave = threadIdx.x >> 5;
  const int lin  = blockIdx.x * 8 + wave;
  const int mt2  = lin & 127;            // 128 token macro-tiles (32 tokens)
  const int nt2  = lin >> 7;             // 96 output macro-tiles (32 outputs)
  const int m  = lane & 15;
  const int hf = lane >> 4;

  const f16* arow0 = xh + (size_t)(mt2 * 32 + m) * D_MODEL;
  const f16* arow1 = arow0 + (size_t)16 * D_MODEL;
  const f16* brow0 = wh + (size_t)(nt2 * 32 + m) * D_MODEL;  // B[k][n] = W[o][k]
  const f16* brow1 = brow0 + (size_t)16 * D_MODEL;

  v8f acc00 = zero8(), acc01 = zero8(), acc10 = zero8(), acc11 = zero8();
  for (int kk = 0; kk < D_MODEL; kk += 32) {
    v16h a0 = frag_f16(arow0 + kk + hf * 8,  arow0 + kk + 16 + hf * 8);
    v16h a1 = frag_f16(arow1 + kk + hf * 8,  arow1 + kk + 16 + hf * 8);
    v16h b0 = frag_f16(brow0 + kk + hf * 8,  brow0 + kk + 16 + hf * 8);
    v16h b1 = frag_f16(brow1 + kk + hf * 8,  brow1 + kk + 16 + hf * 8);
    acc00 = wmma_f16(a0, b0, acc00);
    acc01 = wmma_f16(a0, b1, acc01);
    acc10 = wmma_f16(a1, b0, acc10);
    acc11 = wmma_f16(a1, b1, acc11);
  }
  qkv_store(acc00, mt2 * 2,     nt2 * 2,     m, hf, bias, Q, K, VT);
  qkv_store(acc01, mt2 * 2,     nt2 * 2 + 1, m, hf, bias, Q, K, VT);
  qkv_store(acc10, mt2 * 2 + 1, nt2 * 2,     m, hf, bias, Q, K, VT);
  qkv_store(acc11, mt2 * 2 + 1, nt2 * 2 + 1, m, hf, bias, Q, K, VT);
}

// ---------------------------------------------------------------------------
// Kernel 2: causal flash attention, one wave per (b,h,q-tile of 16).
//   S^T = K . Q^T  (lane owns a fixed q column -> cheap row softmax stats)
//   O^T = V^T . P  (lane owns fixed q -> per-lane alpha / 1/l rescale)
//   att[b][t][h*64+d] (f16) ready as GEMM-A for the projection.
// ---------------------------------------------------------------------------
__global__ __launch_bounds__(256) void attn_kernel(
    const f16* __restrict__ Q, const f16* __restrict__ K,
    const f16* __restrict__ VT, f16* __restrict__ att) {
  __shared__ __align__(16) f16 ldsP[8][16][32];   // per-wave P tile [q][k_local]
  const int lane = threadIdx.x & 31;
  const int wave = threadIdx.x >> 5;
  const int lin  = blockIdx.x * 8 + wave;
  const int qt   = lin & 127;                      // 128 q tiles
  const int bh   = lin >> 7;                       // 0..31
  const int n  = lane & 15;
  const int hf = lane >> 4;

  const f16* Qb = Q  + (size_t)bh * SEQ * HEAD_DIM;
  const f16* Kb = K  + (size_t)bh * SEQ * HEAD_DIM;
  const f16* Vb = VT + (size_t)bh * HEAD_DIM * SEQ;

  // Q as B-fragments (B[d][q]): lane n = q, 16B at d hf*8 and 16+hf*8 per chunk
  const f16* qrow = Qb + (size_t)(qt * 16 + n) * HEAD_DIM;
  v16h bq0 = frag_f16(qrow + hf * 8,      qrow + 16 + hf * 8);        // d 0..31
  v16h bq1 = frag_f16(qrow + 32 + hf * 8, qrow + 48 + hf * 8);        // d 32..63

  v8f accO[4];
#pragma unroll
  for (int t = 0; t < 4; ++t) accO[t] = zero8();
  float m_i = -1e30f, l_i = 0.0f;

  const int qg = qt * 16 + n;
  const int nk = ((qt * 16 + 15) >> 5) + 1;        // causal k-tile count (32 wide)

  for (int kt = 0; kt < nk; ++kt) {
    const int kbase = kt * 32;
    float vals[16];
#pragma unroll
    for (int s = 0; s < 2; ++s) {
      const f16* krow = Kb + (size_t)(kbase + s * 16 + n) * HEAD_DIM; // A row = k
      v16h ak0 = frag_f16(krow + hf * 8,      krow + 16 + hf * 8);    // d 0..31
      v16h ak1 = frag_f16(krow + 32 + hf * 8, krow + 48 + hf * 8);    // d 32..63
      v8f sacc = zero8();
      sacc = wmma_f16(ak0, bq0, sacc);
      sacc = wmma_f16(ak1, bq1, sacc);
#pragma unroll
      for (int j = 0; j < 8; ++j) {                // C row m = k_local = s*16+hf*8+j
        const int kg = kbase + s * 16 + hf * 8 + j;
        vals[s * 8 + j] = (kg <= qg) ? sacc[j] : NEG_FILL;
      }
    }

    // online softmax over k for this lane's q (pair lane^16 holds other 8 rows)
    float tmax = vals[0];
#pragma unroll
    for (int e = 1; e < 16; ++e) tmax = fmaxf(tmax, vals[e]);
    tmax = fmaxf(tmax, __shfl_xor(tmax, 16, 32));
    const float m_new = fmaxf(m_i, tmax);
    float ssum = 0.0f;
    h8 p0, p1;
#pragma unroll
    for (int e = 0; e < 8; ++e) { float p = __expf(vals[e] - m_new);     ssum += p; p0[e] = (f16)p; }
#pragma unroll
    for (int e = 0; e < 8; ++e) { float p = __expf(vals[8 + e] - m_new); ssum += p; p1[e] = (f16)p; }
    ssum += __shfl_xor(ssum, 16, 32);
    const float alpha = __expf(m_i - m_new);
    l_i = l_i * alpha + ssum;
    m_i = m_new;
#pragma unroll
    for (int t = 0; t < 4; ++t)
#pragma unroll
      for (int j = 0; j < 8; ++j) accO[t][j] *= alpha;

    // P: C-layout -> B-layout via wave-private LDS tile [q][k_local]
    *(h8*)&ldsP[wave][n][hf * 8]      = p0;        // k_local  0..15 half
    *(h8*)&ldsP[wave][n][16 + hf * 8] = p1;        // k_local 16..31 half
    __builtin_amdgcn_wave_barrier();               // LDS in-order within wave
    v16h bp = frag_f16(&ldsP[wave][n][hf * 8], &ldsP[wave][n][16 + hf * 8]);

    // O^T += V^T(16d x 32k) x P(32k x 16q), 4 d-tiles
#pragma unroll
    for (int t = 0; t < 4; ++t) {
      const f16* vrow = Vb + (size_t)(t * 16 + n) * SEQ + kbase;       // A row = d
      v16h av = frag_f16(vrow + hf * 8, vrow + 16 + hf * 8);
      accO[t] = wmma_f16(av, bp, accO[t]);
    }
  }

  const float inv = 1.0f / l_i;                    // per-lane (fixed q)
  const int bb = bh >> 4, head = bh & 15;
  f16* orow = att + ((size_t)bb * SEQ + qt * 16 + n) * D_MODEL + head * HEAD_DIM;
#pragma unroll
  for (int t = 0; t < 4; ++t) {                    // rows m = d = t*16+hf*8+j
    h8 ov;
#pragma unroll
    for (int j = 0; j < 8; ++j) ov[j] = (f16)(accO[t][j] * inv);
    *(h8*)(orow + t * 16 + hf * 8) = ov;
  }
}

// ---------------------------------------------------------------------------
// Kernel 3: output projection, 2x2 register-blocked.  out = att.W^T + b (f32)
// ---------------------------------------------------------------------------
__global__ __launch_bounds__(256) void proj_kernel(
    const f16* __restrict__ att, const f16* __restrict__ wh,
    const float* __restrict__ bias, float* __restrict__ out) {
  const int lane = threadIdx.x & 31;
  const int wave = threadIdx.x >> 5;
  const int lin  = blockIdx.x * 8 + wave;
  const int mt2  = lin & 127;            // 128 token macro-tiles
  const int nt2  = lin >> 7;             // 32 output macro-tiles
  const int m  = lane & 15;
  const int hf = lane >> 4;

  const f16* arow0 = att + (size_t)(mt2 * 32 + m) * D_MODEL;
  const f16* arow1 = arow0 + (size_t)16 * D_MODEL;
  const f16* brow0 = wh + (size_t)(nt2 * 32 + m) * D_MODEL;
  const f16* brow1 = brow0 + (size_t)16 * D_MODEL;

  v8f acc00 = zero8(), acc01 = zero8(), acc10 = zero8(), acc11 = zero8();
  for (int kk = 0; kk < D_MODEL; kk += 32) {
    v16h a0 = frag_f16(arow0 + kk + hf * 8,  arow0 + kk + 16 + hf * 8);
    v16h a1 = frag_f16(arow1 + kk + hf * 8,  arow1 + kk + 16 + hf * 8);
    v16h b0 = frag_f16(brow0 + kk + hf * 8,  brow0 + kk + 16 + hf * 8);
    v16h b1 = frag_f16(brow1 + kk + hf * 8,  brow1 + kk + 16 + hf * 8);
    acc00 = wmma_f16(a0, b0, acc00);
    acc01 = wmma_f16(a0, b1, acc01);
    acc10 = wmma_f16(a1, b0, acc10);
    acc11 = wmma_f16(a1, b1, acc11);
  }

  const int o0 = nt2 * 32 + m;
  const int o1 = o0 + 16;
  const float bi0 = bias[o0];
  const float bi1 = bias[o1];
  const size_t r0 = (size_t)(mt2 * 32 + hf * 8) * D_MODEL;
  const size_t r1 = r0 + (size_t)16 * D_MODEL;
#pragma unroll
  for (int j = 0; j < 8; ++j) {
    out[r0 + (size_t)j * D_MODEL + o0] = acc00[j] + bi0;
    out[r0 + (size_t)j * D_MODEL + o1] = acc01[j] + bi1;
    out[r1 + (size_t)j * D_MODEL + o0] = acc10[j] + bi0;
    out[r1 + (size_t)j * D_MODEL + o1] = acc11[j] + bi1;
  }
}

// ---------------------------------------------------------------------------
extern "C" void kernel_launch(void* const* d_in, const int* in_sizes, int n_in,
                              void* d_out, int out_size, void* d_ws, size_t ws_size,
                              hipStream_t stream) {
  (void)in_sizes; (void)n_in; (void)out_size; (void)ws_size;
  const float* x      = (const float*)d_in[0];
  const float* qkv_w  = (const float*)d_in[1];
  const float* qkv_b  = (const float*)d_in[2];
  const float* proj_w = (const float*)d_in[3];
  const float* proj_b = (const float*)d_in[4];
  float* out = (float*)d_out;

  const size_t elems  = (size_t)BATCH * NUM_HEADS * SEQ * HEAD_DIM;  // 4M
  const size_t n_x    = (size_t)TOKENS * D_MODEL;                    // 4M
  const size_t n_wqkv = (size_t)3 * D_MODEL * D_MODEL;               // 3M
  const size_t n_wprj = (size_t)D_MODEL * D_MODEL;                   // 1M

  f16* Q    = (f16*)d_ws;
  f16* K    = Q + elems;
  f16* VT   = K + elems;
  f16* att  = VT + elems;
  f16* xh   = att + elems;
  f16* wqkv = xh + n_x;
  f16* wprj = wqkv + n_wqkv;   // total ws use: 48 MB

  cvt_kernel<<<(int)(n_x / 8 / 256),    256, 0, stream>>>(x,      xh,   (int)(n_x / 8));
  cvt_kernel<<<(int)(n_wqkv / 8 / 256), 256, 0, stream>>>(qkv_w,  wqkv, (int)(n_wqkv / 8));
  cvt_kernel<<<(int)(n_wprj / 8 / 256), 256, 0, stream>>>(proj_w, wprj, (int)(n_wprj / 8));

  qkv_kernel <<<(128 * 96) / 8, 256, 0, stream>>>(xh, wqkv, qkv_b, Q, K, VT);
  attn_kernel<<<(128 * 32) / 8, 256, 0, stream>>>(Q, K, VT, att);
  proj_kernel<<<(128 * 32) / 8, 256, 0, stream>>>(att, wprj, proj_b, out);
}
